// DiPGNN_249108103307
// MI455X (gfx1250) — matrix operand order repaired and checked by reading, not compiled, not run on hardware
//
#include <hip/hip_runtime.h>
#include <hip/hip_bf16.h>
#include <stdint.h>

#define H 128
#define NG 50
#define VARIANCE 0.2f

typedef __attribute__((ext_vector_type(16))) __bf16 v16bf;
typedef __attribute__((ext_vector_type(8)))  float  v8f;

union Frag16 { v16bf v; unsigned int u[8]; uint4 q[2]; };

__device__ __forceinline__ unsigned short f2bf(float f) {
    unsigned int x = __float_as_uint(f);
    x += 0x7fffu + ((x >> 16) & 1u);          // round-to-nearest-even
    return (unsigned short)(x >> 16);
}

__device__ __forceinline__ float swishf(float x) {
    return x * (1.0f / (1.0f + __expf(-x)));
}

__device__ __forceinline__ void atomic_addf(float* p, float v) {
    __hip_atomic_fetch_add(p, v, __ATOMIC_RELAXED, __HIP_MEMORY_SCOPE_AGENT);
}

struct GArgs {
    const float* A0; const float* A1; const float* A2;   // per-K-segment A source
    const int*   i0; const int*   i1; const int*   i2;   // row gather idx (NULL=identity)
    const float* W0; const float* W1; const float* W2;   // per-segment [128,128] fp32 weights
    const float* resid;   // optional: out = resid[row] + swish(acc)
    const int*   scat;    // optional: atomicAdd into out[scat[row]] instead of store
    float* out;
    int M, nseg, do_swish;
};

// C[M,128] = act( sum_seg A_seg[idx_seg[row]] @ W_seg ).
// 8 waves x 16 rows = 128-row tile; per wave: 8 N-tiles x 4 K-steps of
// v_wmma_f32_16x16x32_bf16 (fp32 accumulate). A tile stored fragment-major in
// LDS so every WMMA operand is two contiguous ds_load_b128s.
__global__ __launch_bounds__(256) void k_gemm(GArgs g) {
    // B/weights: [n][k] bf16, 16-half-block XOR swizzle (keeps fragments contiguous)
    __shared__ unsigned short sW[H * H];
    // A: fragment-major [wave-tile(8)][kstep(4)][lane(32)][half(16)]
    __shared__ unsigned short sA[H * H];
    const int tid   = threadIdx.x;
    const int wave  = tid >> 5;
    const int lane  = tid & 31;
    const int mlane = lane & 15;
    const int hi    = lane >> 4;
    const int nblk  = (g.M + 127) >> 7;
    const float* Aseg[3] = { g.A0, g.A1, g.A2 };
    const int*   Iseg[3] = { g.i0, g.i1, g.i2 };
    const float* Wseg[3] = { g.W0, g.W1, g.W2 };

    for (int blk = blockIdx.x; blk < nblk; blk += gridDim.x) {
        const int m0 = blk << 7;
        v8f acc[8];
#pragma unroll
        for (int i = 0; i < 8; ++i)
#pragma unroll
            for (int j = 0; j < 8; ++j) acc[i][j] = 0.0f;

        for (int seg = 0; seg < g.nseg; ++seg) {
            // ---- stage weights: fp32 [k][n] -> bf16 [n][k] (swizzled) ----
            {
                const float* W = Wseg[seg];
                const int t64 = tid << 6;          // 64 contiguous elements per thread
#pragma unroll
                for (int c = 0; c < 64; c += 4) {
                    int i = t64 + c;
                    int k = i >> 7, n0 = i & 127;
                    float4 v = *(const float4*)(W + i);
                    int kblk = k & ~15, klo = k & 15;
                    sW[((n0 + 0) << 7) + (kblk ^ (((n0 + 0) & 7) << 4)) + klo] = f2bf(v.x);
                    sW[((n0 + 1) << 7) + (kblk ^ (((n0 + 1) & 7) << 4)) + klo] = f2bf(v.y);
                    sW[((n0 + 2) << 7) + (kblk ^ (((n0 + 2) & 7) << 4)) + klo] = f2bf(v.z);
                    sW[((n0 + 3) << 7) + (kblk ^ (((n0 + 3) & 7) << 4)) + klo] = f2bf(v.w);
                }
            }
            // ---- stage A tile (gathered rows, fp32 -> bf16, fragment-major) ----
            {
                const float* A = Aseg[seg];
                const int*  ix = Iseg[seg];
                int r  = tid >> 1;
                int c0 = (tid & 1) << 6;
                int row = m0 + r;
                int wt = r >> 4, mr = r & 15;
                const bool ok = (row < g.M);
                long src = 0;
                if (ok) src = ix ? (long)ix[row] : (long)row;
                const float* ap = A + src * H + c0;
#pragma unroll
                for (int c = 0; c < 64; c += 4) {
                    int k  = c0 + c;
                    int ks = k >> 5, kk = k & 31;
                    int h2 = kk >> 4;
                    int halfBase = kk & 15;
                    int ln = mr + (h2 << 4);
                    int base = (((((wt << 2) + ks) << 5) + ln) << 4) + halfBase;
                    uint2 pk; pk.x = 0u; pk.y = 0u;
                    if (ok) {
                        float4 v = *(const float4*)(ap + c);
                        pk.x = (unsigned)f2bf(v.x) | ((unsigned)f2bf(v.y) << 16);
                        pk.y = (unsigned)f2bf(v.z) | ((unsigned)f2bf(v.w) << 16);
                    }
                    *(uint2*)&sA[base] = pk;       // one ds_store_b64
                }
            }
            __syncthreads();

            // ---- WMMA accumulate: 4 K-steps x 8 N-tiles ----
#pragma unroll
            for (int ks = 0; ks < 4; ++ks) {
                Frag16 af;
                {
                    const unsigned short* ap2 =
                        &sA[((((wave << 2) + ks) << 5) + lane) << 4];
                    af.q[0] = *(const uint4*)ap2;          // ds_load_b128
                    af.q[1] = *(const uint4*)(ap2 + 8);    // ds_load_b128
                }
#pragma unroll
                for (int nt = 0; nt < 8; ++nt) {
                    Frag16 bfrag;
                    const int nrow = (nt << 4) + mlane;
                    const int kb = ((ks << 5) + (hi << 4)) ^ ((nrow & 7) << 4);
                    const unsigned short* bp = &sW[(nrow << 7) + kb];
                    bfrag.q[0] = *(const uint4*)bp;
                    bfrag.q[1] = *(const uint4*)(bp + 8);
                    acc[nt] = __builtin_amdgcn_wmma_f32_16x16x32_bf16(
                        false, af.v, false, bfrag.v, (short)0, acc[nt], false, false);
                }
            }
            __syncthreads();
        }

        // ---- epilogue: swish, residual, store / atomic scatter ----
#pragma unroll
        for (int r = 0; r < 8; ++r) {
            int row = m0 + (wave << 4) + r + (hi << 3);
            if (row >= g.M) continue;
#pragma unroll
            for (int nt = 0; nt < 8; ++nt) {
                float v = acc[nt][r];
                if (g.do_swish) v = swishf(v);
                int col = (nt << 4) + mlane;
                if (g.resid) v += g.resid[(size_t)row * H + col];
                if (g.scat) atomic_addf(&g.out[(size_t)g.scat[row] * H + col], v);
                else        g.out[(size_t)row * H + col] = v;
            }
        }
    }
}

// Per-triplet: w = gauss(angle) @ Wsbf computed on the fly (Wsbf cached in LDS),
// msg = m[exp] * m2[ji] * w, atomically reduced into agg[red]. One wave / triplet.
__global__ __launch_bounds__(256) void k_triplet(
    const float* __restrict__ m, const float* __restrict__ m2,
    const float* __restrict__ Wsbf, const float* __restrict__ angle,
    const int* __restrict__ exp_idx, const int* __restrict__ ji_idx,
    const int* __restrict__ red_idx, float* __restrict__ agg,
    int nt, float dmax) {
    __shared__ float sW[NG * H];
    __shared__ float sG[8][NG + 2];
    for (int i = threadIdx.x; i < NG * H; i += 256) sW[i] = Wsbf[i];
    __syncthreads();
    const int wave = threadIdx.x >> 5, lane = threadIdx.x & 31;
    const float step = dmax / (float)(NG - 1);
    const float inv_var = 1.0f / VARIANCE;
    int t = blockIdx.x * 8 + wave;
    const int stride = gridDim.x * 8;
    for (; t < nt; t += stride) {
        float a = angle[t];
        int ei = exp_idx[t], ji = ji_idx[t], ri = red_idx[t];
        if (lane < NG) { float d = a - (float)lane * step; sG[wave][lane] = __expf(-d * d * inv_var); }
        int g1 = lane + 32;
        if (g1 < NG)   { float d = a - (float)g1 * step;   sG[wave][g1] = __expf(-d * d * inv_var); }
        float4 av; av.x = av.y = av.z = av.w = 0.0f;
        const float* wp = sW + (lane << 2);
        for (int g = 0; g < NG; ++g) {
            float e = sG[wave][g];
            float4 wv = *(const float4*)(wp + (g << 7));
            av.x += e * wv.x; av.y += e * wv.y; av.z += e * wv.z; av.w += e * wv.w;
        }
        float4 mi = *(const float4*)(m  + (size_t)ei * H + (lane << 2));
        float4 mj = *(const float4*)(m2 + (size_t)ji * H + (lane << 2));
        float* dst = agg + (size_t)ri * H + (lane << 2);
        atomic_addf(dst + 0, mi.x * mj.x * av.x);
        atomic_addf(dst + 1, mi.y * mj.y * av.y);
        atomic_addf(dst + 2, mi.z * mj.z * av.z);
        atomic_addf(dst + 3, mi.w * mj.w * av.w);
    }
}

// out[e,:] = gauss(x[e]) @ W  (for rbf @ rbf_W). One wave per element.
__global__ __launch_bounds__(256) void k_gaussmv(
    const float* __restrict__ x, const float* __restrict__ W,
    float* __restrict__ out, int n, float dmax) {
    __shared__ float sW[NG * H];
    __shared__ float sG[8][NG + 2];
    for (int i = threadIdx.x; i < NG * H; i += 256) sW[i] = W[i];
    __syncthreads();
    const int wave = threadIdx.x >> 5, lane = threadIdx.x & 31;
    const float step = dmax / (float)(NG - 1);
    const float inv_var = 1.0f / VARIANCE;
    int e = blockIdx.x * 8 + wave;
    const int stride = gridDim.x * 8;
    for (; e < n; e += stride) {
        float a = x[e];
        if (lane < NG) { float d = a - (float)lane * step; sG[wave][lane] = __expf(-d * d * inv_var); }
        int g1 = lane + 32;
        if (g1 < NG)   { float d = a - (float)g1 * step;   sG[wave][g1] = __expf(-d * d * inv_var); }
        float4 av; av.x = av.y = av.z = av.w = 0.0f;
        const float* wp = sW + (lane << 2);
        for (int g = 0; g < NG; ++g) {
            float eg = sG[wave][g];
            float4 wv = *(const float4*)(wp + (g << 7));
            av.x += eg * wv.x; av.y += eg * wv.y; av.z += eg * wv.z; av.w += eg * wv.w;
        }
        *(float4*)(out + (size_t)e * H + (lane << 2)) = av;
    }
}

__global__ void k_fill0(float4* p, long n4) {
    long i = (long)blockIdx.x * blockDim.x + threadIdx.x;
    long s = (long)gridDim.x * blockDim.x;
    float4 z; z.x = z.y = z.z = z.w = 0.0f;
    for (; i < n4; i += s) p[i] = z;
}

__global__ void k_add(float4* d, const float4* s, long n4) {
    long i = (long)blockIdx.x * blockDim.x + threadIdx.x;
    long st = (long)gridDim.x * blockDim.x;
    for (; i < n4; i += st) {
        float4 a = d[i], b = s[i];
        a.x += b.x; a.y += b.y; a.z += b.z; a.w += b.w;
        d[i] = a;
    }
}

__global__ void k_copy(float4* d, const float4* s, long n4) {
    long i = (long)blockIdx.x * blockDim.x + threadIdx.x;
    long st = (long)gridDim.x * blockDim.x;
    for (; i < n4; i += st) d[i] = s[i];
}

__global__ void k_embed(const float* __restrict__ emb, const int* __restrict__ feat,
                        float* __restrict__ atom, int na) {
    long i = (long)blockIdx.x * blockDim.x + threadIdx.x;
    long total = (long)na * (H / 4);
    long st = (long)gridDim.x * blockDim.x;
    for (; i < total; i += st) {
        long a = i >> 5;
        int  c = (int)(i & 31) << 2;
        *(float4*)(atom + a * H + c) = *(const float4*)(emb + (long)feat[a] * H + c);
    }
}

// final: out[a, 0..11] = h[target_idx[a]] @ out_W  (N=12, no activation)
__global__ __launch_bounds__(256) void k_out(const float* __restrict__ h,
                                             const float* __restrict__ Wo,
                                             const int* __restrict__ tgt,
                                             float* __restrict__ out, int na) {
    __shared__ float sw[H * 12];
    for (int i = threadIdx.x; i < H * 12; i += 256) sw[i] = Wo[i];
    __syncthreads();
    long i = (long)blockIdx.x * 256 + threadIdx.x;
    long total = (long)na * 12;
    long st = (long)gridDim.x * 256;
    for (; i < total; i += st) {
        int a = (int)(i / 12), t = (int)(i - (long)a * 12);
        const float* hp = h + (size_t)tgt[a] * H;
        float s = 0.0f;
        for (int k = 0; k < H; ++k) s += hp[k] * sw[k * 12 + t];
        out[i] = s;
    }
}

extern "C" void kernel_launch(void* const* d_in, const int* in_sizes, int n_in,
                              void* d_out, int out_size, void* d_ws, size_t ws_size,
                              hipStream_t stream) {
    (void)in_sizes; (void)out_size; (void)ws_size;
    const int NA_ = 20000, NB_ = 250000, NT_ = 400000;

    // ---- parameter leaves (jax pytree flatten order: dict keys sorted) ----
    // bond_W, efc0, efc1, emb_table, layer0[15], layer1[15], out_W, rbf_W, ro0..ro2
    static const int lsz[39] = {
        49152, 16384, 16384, 12160,
        16384,16384,16384,16384,16384,16384,16384,16384,6400,16384,16384,16384,6400,16384,16384,
        16384,16384,16384,16384,16384,16384,16384,16384,6400,16384,16384,16384,6400,16384,16384,
        1536, 6400, 16384, 16384, 16384 };
    const float* leaf[39];
    int base;
    if (n_in >= 52) {                       // individually flattened leaves
        for (int i = 0; i < 39; ++i) leaf[i] = (const float*)d_in[i];
        base = 39;
    } else {                                 // single packed param blob
        const float* p = (const float*)d_in[0];
        size_t o = 0;
        for (int i = 0; i < 39; ++i) { leaf[i] = p + o; o += (size_t)lsz[i]; }
        base = 1;
    }
    const int*   atom_feat = (const int*)d_in[base + 0];
    const int*   id_i      = (const int*)d_in[base + 1];
    const int*   id_j      = (const int*)d_in[base + 2];
    const float* dist      = (const float*)d_in[base + 3];
    const float* ang_kj    = (const float*)d_in[base + 4];
    const float* ang_im    = (const float*)d_in[base + 5];
    const int*   kj_exp    = (const int*)d_in[base + 6];
    const int*   ji_kj     = (const int*)d_in[base + 7];
    const int*   red_kj    = (const int*)d_in[base + 8];
    const int*   im_exp    = (const int*)d_in[base + 9];
    const int*   ji_im     = (const int*)d_in[base + 10];
    const int*   red_im    = (const int*)d_in[base + 11];
    const int*   tgt       = (const int*)d_in[base + 12];

    const float* bond_W = leaf[0];
    const float* efc0   = leaf[1];
    const float* efc1   = leaf[2];
    const float* emb    = leaf[3];
    const float* out_W  = leaf[34];
    const float* rbf_W  = leaf[35];
    const float* ro[3]  = { leaf[36], leaf[37], leaf[38] };

    // ---- workspace carve ----
    size_t off = 0;
    auto carve = [&](size_t elems) -> float* {
        float* r = (float*)((char*)d_ws + off);
        off += ((elems * 4 + 255) & ~(size_t)255);
        return r;
    };
    float* bond   = carve((size_t)NB_ * H);
    float* mbuf   = carve((size_t)NB_ * H);
    float* m2buf  = carve((size_t)NB_ * H);
    float* agg    = carve((size_t)NB_ * H);
    float* atom   = carve((size_t)NA_ * H);
    float* atomT  = carve((size_t)NA_ * H);
    float* hidden = carve((size_t)NA_ * H);
    float* aggA   = carve((size_t)NA_ * H);

    auto gemm = [&](const float* A0, const int* i0, const float* W0,
                    const float* A1, const int* i1, const float* W1,
                    const float* A2, const int* i2, const float* W2,
                    int nseg, int M, float* out, const float* resid,
                    const int* scat) {
        GArgs a;
        a.A0 = A0; a.A1 = A1; a.A2 = A2;
        a.i0 = i0; a.i1 = i1; a.i2 = i2;
        a.W0 = W0; a.W1 = W1; a.W2 = W2;
        a.resid = resid; a.scat = scat; a.out = out;
        a.M = M; a.nseg = nseg; a.do_swish = 1;
        int nb = (M + 127) >> 7; if (nb > 8192) nb = 8192;
        k_gemm<<<nb, 256, 0, stream>>>(a);
    };
    auto gemm1 = [&](const float* A, const float* W, float* out, int M,
                     const float* resid, const int* scat) {
        gemm(A, nullptr, W, nullptr, nullptr, nullptr, nullptr, nullptr, nullptr,
             1, M, out, resid, scat);
    };

    const long nbond4 = (long)NB_ * H / 4;
    const long natom4 = (long)NA_ * H / 4;

    // 1. atom embedding + 2 FC layers
    k_embed<<<2048, 256, 0, stream>>>(emb, atom_feat, atom, NA_);
    gemm1(atom,  efc0, atomT, NA_, nullptr, nullptr);
    gemm1(atomT, efc1, atom,  NA_, nullptr, nullptr);

    // 2. rbf(dist) @ rbf_W -> agg (reused as the 3rd concat segment)
    k_gaussmv<<<2048, 256, 0, stream>>>(dist, rbf_W, agg, NB_, 5.0f);

    // 3. bond = swish([atom[id_i] | atom[id_j] | rbfW] @ bond_W)
    gemm(atom, id_i, bond_W,
         atom, id_j, bond_W + 128 * H,
         agg,  nullptr, bond_W + 256 * H,
         3, NB_, bond, nullptr, nullptr);

    // 4. hidden = atom
    k_copy<<<1024, 256, 0, stream>>>((float4*)hidden, (const float4*)atom, natom4);

    for (int l = 0; l < 2; ++l) {
        const float* const* Lf = leaf + 4 + 15 * l;
        // layer leaf order: a2b_Wb, a2b_Wi, a2b_Wj, b2a_W1, b2a_W2,
        //                   im{Wm,Wm2,Wout,Wsbf}, kj{Wm,Wm2,Wout,Wsbf}, res0, res1
        const float* dWm[2]   = { Lf[9],  Lf[5] };   // kj, im
        const float* dWm2[2]  = { Lf[10], Lf[6] };
        const float* dWout[2] = { Lf[11], Lf[7] };
        const float* dWsbf[2] = { Lf[12], Lf[8] };
        const float* dang[2]  = { ang_kj, ang_im };
        const int* dexp[2] = { kj_exp, im_exp };
        const int* dji[2]  = { ji_kj,  ji_im  };
        const int* dred[2] = { red_kj, red_im };

        // --- direction kj (bond is pre-update state) ---
        gemm1(bond, dWm[0],  mbuf,  NB_, nullptr, nullptr);
        gemm1(bond, dWm2[0], m2buf, NB_, nullptr, nullptr);
        k_fill0<<<4096, 256, 0, stream>>>((float4*)agg, nbond4);
        k_triplet<<<4096, 256, 0, stream>>>(mbuf, m2buf, dWsbf[0], dang[0],
                                            dexp[0], dji[0], dred[0], agg, NT_, 3.14f);
        gemm1(agg, dWout[0], agg, NB_, nullptr, nullptr);   // delta_kj (in place)

        // --- direction im: compute m/m2 from OLD bond before applying delta_kj ---
        gemm1(bond, dWm[1],  mbuf,  NB_, nullptr, nullptr);
        gemm1(bond, dWm2[1], m2buf, NB_, nullptr, nullptr);
        k_add<<<4096, 256, 0, stream>>>((float4*)bond, (const float4*)agg, nbond4);
        k_fill0<<<4096, 256, 0, stream>>>((float4*)agg, nbond4);
        k_triplet<<<4096, 256, 0, stream>>>(mbuf, m2buf, dWsbf[1], dang[1],
                                            dexp[1], dji[1], dred[1], agg, NT_, 3.14f);
        gemm1(agg, dWout[1], agg, NB_, nullptr, nullptr);
        k_add<<<4096, 256, 0, stream>>>((float4*)bond, (const float4*)agg, nbond4);

        // --- residual FCs ---
        gemm1(bond, Lf[13], bond, NB_, bond, nullptr);
        gemm1(bond, Lf[14], bond, NB_, bond, nullptr);

        // --- Bond2Atom: fused scatter-add epilogue, then atom update ---
        k_fill0<<<1024, 256, 0, stream>>>((float4*)aggA, natom4);
        gemm1(bond, Lf[3], aggA, NB_, nullptr, id_i);       // aggA[id_i] += swish(bond@W1)
        gemm1(aggA, Lf[4], atom, NA_, atom, nullptr);       // atom += swish(aggA@W2)

        // --- Atom2Bond: 3-segment gathered GEMM with bond residual ---
        gemm(atom, id_i, Lf[1],          // a2b_Wi
             atom, id_j, Lf[2],          // a2b_Wj
             bond, nullptr, Lf[0],       // a2b_Wb
             3, NB_, bond, bond, nullptr);

        // --- hidden += atom ---
        k_add<<<1024, 256, 0, stream>>>((float4*)hidden, (const float4*)atom, natom4);
    }

    // readout (3 swish FCs) + final projection with target gather
    gemm1(hidden, ro[0], atomT, NA_, nullptr, nullptr);
    gemm1(atomT,  ro[1], atom,  NA_, nullptr, nullptr);
    gemm1(atom,   ro[2], atomT, NA_, nullptr, nullptr);
    k_out<<<1024, 256, 0, stream>>>(atomT, out_W, tgt, (float*)d_out, NA_);
}
